// Model_51402168599311
// MI455X (gfx1250) — matrix operand order repaired and checked by reading, not compiled, not run on hardware
//
#include <hip/hip_runtime.h>
#include <hip/hip_bf16.h>
#include <cmath>

// ---------------------------------------------------------------------------
// Shapes
#define Bb 16
#define Tt 288
#define Nn 1024
#define Dd 32
#define Mm 4096
#define KTOP 8
#define Hh 4
#define HD 8
#define ND (Nn*Dd)        // 32768
#define NT 289            // conv output length
#define NT_PAD 320        // padded to 20 tiles of 16 (uniform per-wave work)
#define NTILES 20

typedef __attribute__((ext_vector_type(16))) _Float16 v16h;
typedef __attribute__((ext_vector_type(8)))  _Float16 v8h;
typedef __attribute__((ext_vector_type(4)))  _Float16 v4h;
typedef __attribute__((ext_vector_type(8)))  float    v8f;
typedef __attribute__((ext_vector_type(4)))  float    v4f_;
typedef __attribute__((ext_vector_type(4)))  int      v4i_;

// CDNA5 async VMEM->LDS path (ASYNCcnt-tracked, no VGPR round trip).
#if __has_builtin(__builtin_amdgcn_global_load_async_to_lds_b128) && \
    __has_builtin(__builtin_amdgcn_s_wait_asynccnt)
#define USE_ASYNC_LDS 1
#else
#define USE_ASYNC_LDS 0
#endif

#if USE_ASYNC_LDS
__device__ __forceinline__ void async_g2l_16(const void* g, void* l) {
    // Builtin signature: (AS1 int4*, AS3 int4*, imm offset, imm cpol).
    // Generic->AS casts via integers: global generic ptr bits == AS1 bits;
    // LDS generic ptr low 32 bits == LDS byte offset (ISA aperture mapping).
    __builtin_amdgcn_global_load_async_to_lds_b128(
        (__attribute__((address_space(1))) v4i_*)(uintptr_t)g,
        (__attribute__((address_space(3))) v4i_*)(unsigned)(uintptr_t)l,
        0, 0);
}
#endif

__device__ __forceinline__ float gelu_f(float x) {
    return 0.5f * x * (1.0f + erff(x * 0.70710678118654752f));
}

// ---------------------------------------------------------------------------
// Kernel 1: convert pw_w (32768x1024 fp32, row-major, c contiguous) -> f16
__global__ __launch_bounds__(256) void k_cvt_w(const float* __restrict__ w,
                                               _Float16* __restrict__ w16, int n4) {
    int i = blockIdx.x * 256 + threadIdx.x;
    if (i < n4) {
        v4f_ v = *(const v4f_*)(w + (size_t)i * 4);
        v4h h;
        for (int e = 0; e < 4; ++e) h[e] = (_Float16)v[e];
        *(v4h*)(w16 + (size_t)i * 4) = h;
    }
}

// ---------------------------------------------------------------------------
// Kernel 2: depthwise conv (k=12, pad 6/6) + bias + GELU -> y16[b][t'][c] f16
__global__ __launch_bounds__(256) void k_conv(const float* __restrict__ x,
                                              const float* __restrict__ dw_w,
                                              const float* __restrict__ dw_b,
                                              _Float16* __restrict__ y16) {
    int bt = blockIdx.x;           // b*NT_PAD + t'
    int b  = bt / NT_PAD;
    int tp = bt % NT_PAD;
    for (int c = threadIdx.x; c < Nn; c += 256) {
        _Float16 r = (_Float16)0.0f;
        if (tp < NT) {
            float acc = dw_b[c];
            #pragma unroll
            for (int i = 0; i < 12; ++i) {
                int t = tp + i - 6;
                if (t >= 0 && t < Tt)
                    acc += x[((size_t)b * Tt + t) * Nn + c] * dw_w[c * 12 + i];
            }
            r = (_Float16)gelu_f(acc);
        }
        y16[(size_t)bt * Nn + c] = r;
    }
}

// ---------------------------------------------------------------------------
// Kernel 3: big GEMM (per batch: W[32768x1024] * Y[1024x320]) fused with
// bias + GELU + time-mean -> q_raw[b][o].
// Block = 64 rows x 320 cols; 8 waves = 4 row strips x 2 col groups, each
// wave 16 rows x 10 tiles.  K streamed in 32-chunks through double-buffered
// LDS; staging uses GLOBAL_LOAD_ASYNC_TO_LDS so the wave issues the next
// chunk's copies and falls straight into this chunk's 10 WMMAs, closing the
// pipe with a single s_wait_asynccnt 0 before the barrier.
__global__ __launch_bounds__(256) void k_gemm_mean(const _Float16* __restrict__ w16,
                                                   const _Float16* __restrict__ y16,
                                                   const float* __restrict__ pw_b,
                                                   float* __restrict__ q_raw) {
    __shared__ _Float16 sA[2][64 * 32];      // [buf][row][k]
    __shared__ _Float16 sB[2][NT_PAD * 32];  // [buf][t][k]
    __shared__ float red[4][2][16];

    int b    = blockIdx.y;
    int rg   = blockIdx.x;                 // row group of 64 (0..511)
    int tid  = threadIdx.x;
    int lane = tid & 31, wave = tid >> 5;
    int strip = wave >> 1;                 // 0..3
    int grp   = wave & 1;                  // 0..1
    int rowblk  = rg * 64;
    int rowbase = rowblk + strip * 16;
    int tile0   = grp * 10;

    const _Float16* wbase = w16 + (size_t)rowblk * Nn;
    const _Float16* ybase = y16 + (size_t)b * NT_PAD * Nn;

    v8f acc[10];
    for (int j = 0; j < 10; ++j)
        for (int e = 0; e < 8; ++e) acc[j][e] = 0.0f;

    // fragment addressing (contiguous k = 16h + s for both A and B)
    int arow = strip * 16 + (lane & 15);
    int akh  = (lane >> 4) * 16;
    int bcol = (lane & 15);
    int bkh  = (lane >> 4) * 16;

    // staging unit decomposition: 16B units; A = 256 units, B = 1280 units
    int uaR = tid >> 2, uaS = (tid & 3) * 8;          // A: 1 unit/thread

    // ---- stage chunk 0
    {
#if USE_ASYNC_LDS
        async_g2l_16(wbase + (size_t)uaR * Nn + uaS, &sA[0][uaR * 32 + uaS]);
        #pragma unroll
        for (int i = 0; i < 5; ++i) {
            int u = tid + 256 * i, t = u >> 2, s = (u & 3) * 8;
            async_g2l_16(ybase + (size_t)t * Nn + s, &sB[0][t * 32 + s]);
        }
        __builtin_amdgcn_s_wait_asynccnt(0);
#else
        for (int u = tid; u < 128; u += 256) {
            int r = u >> 1, h = (u & 1) * 16;
            *(v16h*)&sA[0][r * 32 + h] = *(const v16h*)(wbase + (size_t)r * Nn + h);
        }
        for (int u = tid; u < 640; u += 256) {
            int t = u >> 1, h = (u & 1) * 16;
            *(v16h*)&sB[0][t * 32 + h] = *(const v16h*)(ybase + (size_t)t * Nn + h);
        }
#endif
    }
    __syncthreads();

    for (int chunk = 0; chunk < 32; ++chunk) {
        int cur = chunk & 1;
        // stage next chunk into the other buffer (overlaps with WMMAs below)
        if (chunk + 1 < 32) {
            int c0 = (chunk + 1) * 32;
#if USE_ASYNC_LDS
            async_g2l_16(wbase + (size_t)uaR * Nn + c0 + uaS,
                         &sA[cur ^ 1][uaR * 32 + uaS]);
            #pragma unroll
            for (int i = 0; i < 5; ++i) {
                int u = tid + 256 * i, t = u >> 2, s = (u & 3) * 8;
                async_g2l_16(ybase + (size_t)t * Nn + c0 + s,
                             &sB[cur ^ 1][t * 32 + s]);
            }
#else
            for (int u = tid; u < 128; u += 256) {
                int r = u >> 1, h = (u & 1) * 16;
                *(v16h*)&sA[cur ^ 1][r * 32 + h] =
                    *(const v16h*)(wbase + (size_t)r * Nn + c0 + h);
            }
            for (int u = tid; u < 640; u += 256) {
                int t = u >> 1, h = (u & 1) * 16;
                *(v16h*)&sB[cur ^ 1][t * 32 + h] =
                    *(const v16h*)(ybase + (size_t)t * Nn + c0 + h);
            }
#endif
        }
        // compute on current buffer; B-fragment loads pipelined one tile ahead
        v8h alo = *(const v8h*)&sA[cur][arow * 32 + akh];
        v8h ahi = *(const v8h*)&sA[cur][arow * 32 + akh + 8];
        v16h a;
        #pragma unroll
        for (int e = 0; e < 8; ++e) { a[e] = alo[e]; a[e + 8] = ahi[e]; }

        v16h bf = *(const v16h*)&sB[cur][(tile0 * 16 + bcol) * 32 + bkh];
        #pragma unroll
        for (int j = 0; j < 10; ++j) {
            v16h bfn;
            if (j < 9) {
                int t = (tile0 + j + 1) * 16 + bcol;
                bfn = *(const v16h*)&sB[cur][t * 32 + bkh];
            }
            acc[j] = __builtin_amdgcn_wmma_f32_16x16x32_f16(
                false, a, false, bf, (short)0, acc[j], false, false);
            bf = bfn;
        }
#if USE_ASYNC_LDS
        __builtin_amdgcn_s_wait_asynccnt(0);
#endif
        __syncthreads();
    }

    // Epilogue: bias + GELU, mask t>=289, partial time-sum per lane
    int rofs = 8 * (lane >> 4);
    float bias[8], s[8];
    #pragma unroll
    for (int i = 0; i < 8; ++i) { bias[i] = pw_b[rowbase + rofs + i]; s[i] = 0.0f; }
    #pragma unroll
    for (int j = 0; j < 10; ++j) {
        int t = (tile0 + j) * 16 + bcol;
        if (t < NT) {
            #pragma unroll
            for (int i = 0; i < 8; ++i) s[i] += gelu_f(acc[j][i] + bias[i]);
        }
    }
    #pragma unroll
    for (int m = 1; m < 16; m <<= 1)
        #pragma unroll
        for (int i = 0; i < 8; ++i) s[i] += __shfl_xor(s[i], m, 16);

    if ((lane & 15) == 0) {
        #pragma unroll
        for (int i = 0; i < 8; ++i) red[strip][grp][rofs + i] = s[i];
    }
    __syncthreads();
    if (tid < 64) {
        int st = tid >> 4, row = tid & 15;
        float tot = red[st][0][row] + red[st][1][row];
        q_raw[(size_t)b * ND + rowblk + st * 16 + row] = tot * (1.0f / 289.0f);
    }
}

// ---------------------------------------------------------------------------
// Kernel 4: LayerNorm over D=32 per (b,n); one wave per row (wave32 == D).
__global__ __launch_bounds__(256) void k_ln(const float* __restrict__ q_raw,
                                            const float* __restrict__ g,
                                            const float* __restrict__ bb,
                                            float* __restrict__ q_ws,
                                            float* __restrict__ out_q) {
    int wid  = (int)((blockIdx.x * 256 + threadIdx.x) >> 5);
    int lane = threadIdx.x & 31;
    float v  = q_raw[(size_t)wid * 32 + lane];
    float mu = v;
    #pragma unroll
    for (int m = 1; m < 32; m <<= 1) mu += __shfl_xor(mu, m, 32);
    mu *= (1.0f / 32.0f);
    float d = v - mu, var = d * d;
    #pragma unroll
    for (int m = 1; m < 32; m <<= 1) var += __shfl_xor(var, m, 32);
    var *= (1.0f / 32.0f);
    float o = d * rsqrtf(var + 1e-5f) * g[lane] + bb[lane];
    q_ws[(size_t)wid * 32 + lane]  = o;
    out_q[(size_t)wid * 32 + lane] = o;
}

// ---------------------------------------------------------------------------
// Kernel 5: per-batch L2 norm of q (32768 elements)
__global__ __launch_bounds__(256) void k_qnorm(const float* __restrict__ q_ws,
                                               float* __restrict__ qn) {
    int b = blockIdx.x;
    float ss = 0.0f;
    for (int i = threadIdx.x; i < ND; i += 256) {
        float v = q_ws[(size_t)b * ND + i];
        ss += v * v;
    }
    #pragma unroll
    for (int m = 1; m < 32; m <<= 1) ss += __shfl_xor(ss, m, 32);
    __shared__ float red[8];
    int lane = threadIdx.x & 31, wave = threadIdx.x >> 5;
    if (lane == 0) red[wave] = ss;
    __syncthreads();
    if (threadIdx.x == 0) {
        float t = 0.0f;
        for (int w = 0; w < 8; ++w) t += red[w];
        qn[b] = sqrtf(t);
    }
}

// ---------------------------------------------------------------------------
// Kernel 6: similarity GEMM (16 x 4096, K=32768) via WMMA f16, fused with
// memory-row L2 norms, season mask and time-diversity scaling.  HBM-bound:
// the 512 MB bank is streamed exactly once (~22us at 23.3 TB/s), register
// double-buffered with global_prefetch running ahead.
__global__ __launch_bounds__(512) void k_sim(const float* __restrict__ q_ws,
                                             const float* __restrict__ mem,
                                             const float* __restrict__ qn,
                                             const int* __restrict__ season_q,
                                             const int* __restrict__ mem_seasons,
                                             const float* __restrict__ year_q,
                                             const float* __restrict__ mem_years,
                                             float* __restrict__ sim) {
    int mt = blockIdx.x;                   // 0..255
    int tid = threadIdx.x, lane = tid & 31, wave = tid >> 5;   // 16 waves
    int col = lane & 15;
    int m   = mt * 16 + col;
    const float* mrow = mem  + (size_t)m   * ND + ((lane >> 4) * 16);
    const float* qrow = q_ws + (size_t)col * ND + ((lane >> 4) * 16);

    v8f acc;
    for (int e = 0; e < 8; ++e) acc[e] = 0.0f;
    float ssq = 0.0f;
    const int KC = ND / 16;                // 2048 per wave
    int k0 = wave * KC;

    v4f_ mcur[4];
    {
        const v4f_* mb = (const v4f_*)(mrow + k0);
        #pragma unroll
        for (int u = 0; u < 4; ++u) mcur[u] = mb[u];
    }

    for (int it = 0; it < KC / 32; ++it) {
        int c = k0 + it * 32;
        v4f_ mnxt[4];
        if (it + 1 < KC / 32) {
            const v4f_* mb = (const v4f_*)(mrow + c + 32);
            #pragma unroll
            for (int u = 0; u < 4; ++u) mnxt[u] = mb[u];
            __builtin_prefetch(mrow + c + 512, 0, 1);   // global_prefetch ahead
        }
        const float* qa = qrow + c;
        v16h a;
        #pragma unroll
        for (int e = 0; e < 8; ++e) {
            a[e]     = (_Float16)qa[e];
            a[e + 8] = (_Float16)qa[8 + e];
        }
        v16h bf;
        float lss = 0.0f;
        #pragma unroll
        for (int u = 0; u < 4; ++u) {
            #pragma unroll
            for (int e = 0; e < 4; ++e) {
                float x = mcur[u][e];
                bf[u * 4 + e] = (_Float16)x;
                lss += x * x;
            }
        }
        ssq += lss;
        acc = __builtin_amdgcn_wmma_f32_16x16x32_f16(
            false, a, false, bf, (short)0, acc, false, false);
        #pragma unroll
        for (int u = 0; u < 4; ++u) mcur[u] = mnxt[u];
    }

    __shared__ float lacc[16][256];
    __shared__ float lssq[16][32];
    #pragma unroll
    for (int e = 0; e < 8; ++e) lacc[wave][lane * 8 + e] = acc[e];
    lssq[wave][lane] = ssq;
    __syncthreads();

    if (wave == 0) {
        float c8[8];
        #pragma unroll
        for (int e = 0; e < 8; ++e) {
            float t = 0.0f;
            for (int w = 0; w < 16; ++w) t += lacc[w][lane * 8 + e];
            c8[e] = t;
        }
        float msq = 0.0f;
        for (int w = 0; w < 16; ++w) msq += lssq[w][col] + lssq[w][col + 16];
        float mn = fmaxf(sqrtf(msq), 1e-12f);
        int   sm = mem_seasons[m];
        float ym = mem_years[m];
        #pragma unroll
        for (int e = 0; e < 8; ++e) {
            int b = e + 8 * (lane >> 4);
            float qv = fmaxf(qn[b], 1e-12f);
            float sv = c8[e] / (qv * mn);
            if (season_q[b] != sm) sv = -10000.0f;
            float dy  = fabsf(year_q[b] - ym);
            float div = 1.0f - __expf(-dy * 0.5f);   // TAU_TIME = 2
            sv *= (0.5f + 0.5f * div);
            sim[(size_t)b * Mm + m] = sv;
        }
    }
}

// ---------------------------------------------------------------------------
// Kernel 7: top-8 of 4096 per batch (lax.top_k tie-breaking: earliest index).
__global__ __launch_bounds__(256) void k_topk(const float* __restrict__ sim,
                                              int* __restrict__ idx) {
    int b = blockIdx.x, t = threadIdx.x;
    float vals[16];
    unsigned mask = 0;
    #pragma unroll
    for (int j = 0; j < 16; ++j) vals[j] = sim[(size_t)b * Mm + t * 16 + j];
    __shared__ float bv[256];
    __shared__ int   bi[256];
    __shared__ int   chosen;
    for (int p = 0; p < KTOP; ++p) {
        float best = -3.4e38f; int bj = 0;
        #pragma unroll
        for (int j = 0; j < 16; ++j)
            if (!((mask >> j) & 1u) && vals[j] > best) { best = vals[j]; bj = j; }
        bv[t] = best; bi[t] = t * 16 + bj;
        __syncthreads();
        if (t == 0) {
            float gb = -3.4e38f; int gi = 0;
            for (int u = 0; u < 256; ++u)
                if (bv[u] > gb) { gb = bv[u]; gi = bi[u]; }
            idx[b * KTOP + p] = gi;
            chosen = gi;
        }
        __syncthreads();
        if ((chosen >> 4) == t) mask |= 1u << (chosen & 15);
        __syncthreads();
    }
}

// ---------------------------------------------------------------------------
// Kernel 8: per-(b,n) cross attention (32x32 matvecs, H=4 heads of 8) + out
// projection + final projection.  One wave per token; weights staged in LDS.
__global__ __launch_bounds__(256) void k_attn(const float* __restrict__ q_ws,
                                              const float* __restrict__ mem,
                                              const int* __restrict__ idx,
                                              const float* __restrict__ in_w,
                                              const float* __restrict__ in_b,
                                              const float* __restrict__ out_w,
                                              const float* __restrict__ out_b,
                                              const float* __restrict__ proj_w,
                                              const float* __restrict__ proj_b,
                                              float* __restrict__ out) {
    __shared__ float lw[5 * 1024];   // Wq | Wk | Wv | Wo | Wp
    __shared__ float lb[160];        // bq bk bv (96) | bo (32) | bp (32)
    int tid = threadIdx.x;
    for (int i = tid; i < 3072; i += 256) lw[i]        = in_w[i];
    for (int i = tid; i < 1024; i += 256) lw[3072 + i] = out_w[i];
    for (int i = tid; i < 1024; i += 256) lw[4096 + i] = proj_w[i];
    if (tid < 96) lb[tid] = in_b[tid];
    if (tid < 32) { lb[96 + tid] = out_b[tid]; lb[128 + tid] = proj_b[tid]; }
    __syncthreads();

    int lane = tid & 31, wave = tid >> 5;
    int p = blockIdx.x * 8 + wave;        // (b,n) pair, 0..16383
    int b = p >> 10, n = p & 1023;

    float qv = q_ws[(size_t)p * 32 + lane];
    float qp = lb[lane];
    for (int j = 0; j < 32; ++j) qp += lw[lane * 32 + j] * __shfl(qv, j, 32);

    float kp[KTOP], vp[KTOP];
    for (int r = 0; r < KTOP; ++r) {
        int mi = idx[b * KTOP + r];
        float kvv = mem[(size_t)mi * ND + n * 32 + lane];
        float kk = lb[32 + lane], vv = lb[64 + lane];
        for (int j = 0; j < 32; ++j) {
            float c = __shfl(kvv, j, 32);
            kk += lw[1024 + lane * 32 + j] * c;
            vv += lw[2048 + lane * 32 + j] * c;
        }
        kp[r] = kk; vp[r] = vv;
    }

    float sc[KTOP];
    const float scale = 0.35355339059327373f;  // 1/sqrt(HD)
    #pragma unroll
    for (int r = 0; r < KTOP; ++r) {
        float pp = qp * kp[r];
        #pragma unroll
        for (int o = 4; o; o >>= 1) pp += __shfl_xor(pp, o, 8);
        sc[r] = pp * scale;
    }
    float mx = sc[0];
    #pragma unroll
    for (int r = 1; r < KTOP; ++r) mx = fmaxf(mx, sc[r]);
    float den = 0.0f;
    #pragma unroll
    for (int r = 0; r < KTOP; ++r) { sc[r] = __expf(sc[r] - mx); den += sc[r]; }
    float inv = 1.0f / den;
    float o1 = 0.0f;
    #pragma unroll
    for (int r = 0; r < KTOP; ++r) o1 += sc[r] * inv * vp[r];

    float o2 = lb[96 + lane];
    for (int j = 0; j < 32; ++j) o2 += lw[3072 + lane * 32 + j] * __shfl(o1, j, 32);
    float o3 = lb[128 + lane];
    for (int j = 0; j < 32; ++j) o3 += lw[4096 + lane * 32 + j] * __shfl(o2, j, 32);

    out[(size_t)p * 32 + lane] = o3;
}

// ---------------------------------------------------------------------------
extern "C" void kernel_launch(void* const* d_in, const int* in_sizes, int n_in,
                              void* d_out, int out_size, void* d_ws, size_t ws_size,
                              hipStream_t stream) {
    const float* x           = (const float*)d_in[0];
    const float* year_q      = (const float*)d_in[1];
    const float* dw_w        = (const float*)d_in[2];
    const float* dw_b        = (const float*)d_in[3];
    const float* pw_w        = (const float*)d_in[4];
    const float* pw_bv       = (const float*)d_in[5];
    const float* ln_g        = (const float*)d_in[6];
    const float* ln_bv       = (const float*)d_in[7];
    const float* mem         = (const float*)d_in[8];
    const float* mem_years   = (const float*)d_in[9];
    const float* in_w        = (const float*)d_in[10];
    const float* in_b        = (const float*)d_in[11];
    const float* out_w       = (const float*)d_in[12];
    const float* out_b       = (const float*)d_in[13];
    const float* proj_w      = (const float*)d_in[14];
    const float* proj_b      = (const float*)d_in[15];
    const int*   season_q    = (const int*)d_in[16];
    const int*   mem_seasons = (const int*)d_in[17];
    float* out = (float*)d_out;

    char* ws = (char*)d_ws;
    _Float16* w16 = (_Float16*)ws;                                  // 64 MB
    _Float16* y16 = (_Float16*)(ws + ((size_t)64 << 20));           // 10 MB
    float* q_raw  = (float*)(ws + ((size_t)64 << 20) + ((size_t)10 << 20) + ((size_t)1 << 20));
    float* q_ws   = q_raw + (size_t)Bb * ND;                        // 2 MB
    float* qn     = q_ws  + (size_t)Bb * ND;                        // 64 B
    float* sim    = qn + 64;                                        // 256 KB
    int*   idx    = (int*)(sim + (size_t)Bb * Mm);

    k_cvt_w<<<(ND * Nn / 4 + 255) / 256, 256, 0, stream>>>(pw_w, w16, ND * Nn / 4);
    k_conv<<<Bb * NT_PAD, 256, 0, stream>>>(x, dw_w, dw_b, y16);

    dim3 gg(ND / 64, Bb);
    k_gemm_mean<<<gg, 256, 0, stream>>>(w16, y16, pw_bv, q_raw);

    k_ln<<<(Bb * Nn) / 8, 256, 0, stream>>>(q_raw, ln_g, ln_bv, q_ws,
                                            out + (size_t)Bb * Nn * Dd);
    k_qnorm<<<Bb, 256, 0, stream>>>(q_ws, qn);
    k_sim<<<Mm / 16, 512, 0, stream>>>(q_ws, mem, qn, season_q, mem_seasons,
                                       year_q, mem_years, sim);
    k_topk<<<Bb, 256, 0, stream>>>(sim, idx);
    k_attn<<<(Bb * Nn) / 8, 256, 0, stream>>>(q_ws, mem, idx, in_w, in_b,
                                              out_w, out_b, proj_w, proj_b, out);
}
